// GOGNNReason_68015102099913
// MI455X (gfx1250) — compile-verified, matrix operand order checked by reading, not run on hardware
//
#include <hip/hip_runtime.h>
#include <hip/hip_bf16.h>

#define DEV __device__ __forceinline__

typedef __bf16 bf16_t;
typedef __attribute__((ext_vector_type(16))) __bf16 v16bf;
typedef __attribute__((ext_vector_type(8)))  __bf16 v8bf;
typedef __attribute__((ext_vector_type(8)))  float  v8f;

// ---------------- problem constants ----------------
constexpr int B_IMG     = 32;
constexpr int NPER      = 40;
constexpr int NNODE     = B_IMG * NPER;           // 1280
constexpr int PAIRS_PER = NPER * (NPER - 1);      // 1560
constexpr int NPAIR     = B_IMG * PAIRS_PER;      // 49920
constexpr int OBJ_DIM   = 4096;
constexpr int HID       = 512;
constexpr int OUT_DIM   = 512;
constexpr int NCLS      = 151;
constexpr int NCLS_PAD  = 256;                    // 16 n-tiles -> TPW=2, branchless
constexpr int TSTEPS    = 3;

// ---------------- WMMA fragment helpers ----------------
// bf16 16x16x32 fragment: lane 0..15 -> row lane&15, K = kb..kb+7, 16+kb..16+kb+7
// with kb = (lane>=16)*8. B identical with col = lane&15 when B is stored
// transposed (Wt row-major [N,K]).
DEV v16bf load_frag16(const __hip_bfloat16* p) {
  const v8bf* q = (const v8bf*)p;     // p already includes row base + kb + k0
  v8bf lo = q[0];                     // K = k0+kb .. +7
  v8bf hi = q[2];                     // K = k0+kb+16 .. +23
  return __builtin_shufflevector(lo, hi, 0,1,2,3,4,5,6,7,8,9,10,11,12,13,14,15);
}

DEV v8f wmma_bf16(v16bf a, v16bf b, v8f c) {
  return __builtin_amdgcn_wmma_f32_16x16x32_bf16(
      /*neg_a=*/false, a, /*neg_b=*/false, b,
      /*c_mod=*/(short)0, c, /*reuse_a=*/false, /*reuse_b=*/false);
}

DEV float fast_sigmoid(float x) { return 1.0f / (1.0f + __expf(-x)); }

// ---------------- conversion kernels ----------------
__global__ __launch_bounds__(256) void k_f32_to_bf16(const float* __restrict__ in,
                                                     __hip_bfloat16* __restrict__ out,
                                                     int n) {
  int i = blockIdx.x * 256 + threadIdx.x;
  if (i < n) out[i] = __float2bfloat16(in[i]);
}

// W [K,N] f32 -> Wt [Npad,K] bf16 (rows >= N zero-filled)
__global__ __launch_bounds__(256) void k_transpose_bf16(const float* __restrict__ W,
                                                        __hip_bfloat16* __restrict__ Wt,
                                                        int K, int N, int Npad) {
  int i = blockIdx.x * 256 + threadIdx.x;
  if (i >= Npad * K) return;
  int nrow = i / K, k = i - nrow * K;
  float v = (nrow < N) ? W[(size_t)k * N + nrow] : 0.0f;
  Wt[i] = __float2bfloat16(v);
}

// ---------------- generic dual-source WMMA GEMM (branchless) ----------------
// C[m0+16, N] = act( A1 @ B1^T (+ A2 @ B2^T) + bias )
// A row-major [M,K] bf16; B stored transposed bf16, physically 128*TPW rows.
// Block = 256 threads = 8 waves; wave w owns n-tiles nt = w + 8*t, t<TPW.
template <int TPW>
__global__ __launch_bounds__(256) void k_gemm(
    const __hip_bfloat16* __restrict__ A1, int lda1,
    const __hip_bfloat16* __restrict__ B1, int ldb1, int K1,
    const __hip_bfloat16* __restrict__ A2, int lda2,
    const __hip_bfloat16* __restrict__ B2, int ldb2, int K2,
    const float* __restrict__ bias, int N, int act,
    float* __restrict__ Cf, __hip_bfloat16* __restrict__ Cb, int ldc) {
  const int lane = threadIdx.x & 31;
  const int wave = threadIdx.x >> 5;
  const int m0   = blockIdx.x * 16;
  const int kb   = (lane >> 4) * 8;
  const int colb = lane & 15;
  const int arow = m0 + (lane & 15);

  v8f acc[TPW];
#pragma unroll
  for (int t = 0; t < TPW; ++t) acc[t] = v8f{};

  {
    const __hip_bfloat16* Ar = A1 + (size_t)arow * lda1 + kb;
    for (int k0 = 0; k0 < K1; k0 += 32) {
      v16bf a = load_frag16(Ar + k0);
#pragma unroll
      for (int t = 0; t < TPW; ++t) {
        const int nt = wave + 8 * t;           // strided: always < 8*TPW tiles
        const __hip_bfloat16* Br = B1 + (size_t)(nt * 16 + colb) * ldb1 + kb + k0;
        acc[t] = wmma_bf16(a, load_frag16(Br), acc[t]);
      }
    }
  }
  if (A2 != nullptr) {                          // scalar (kernel-arg) branch
    const __hip_bfloat16* Ar = A2 + (size_t)arow * lda2 + kb;
    for (int k0 = 0; k0 < K2; k0 += 32) {
      v16bf a = load_frag16(Ar + k0);
#pragma unroll
      for (int t = 0; t < TPW; ++t) {
        const int nt = wave + 8 * t;
        const __hip_bfloat16* Br = B2 + (size_t)(nt * 16 + colb) * ldb2 + kb + k0;
        acc[t] = wmma_bf16(a, load_frag16(Br), acc[t]);
      }
    }
  }

  const int rowb = m0 + 8 * (lane >> 4);        // C layout: row = rowb + v
#pragma unroll
  for (int t = 0; t < TPW; ++t) {
    const int col = (wave + 8 * t) * 16 + colb;
    if (col >= N) continue;                     // store-only guard (no WMMA inside)
    float bv = bias ? bias[col] : 0.0f;
#pragma unroll
    for (int v = 0; v < 8; ++v) {
      float x = acc[t][v] + bv;
      if (act == 1) x = fmaxf(x, 0.0f);
      if (Cf) Cf[(size_t)(rowb + v) * ldc + col] = x;
      if (Cb) Cb[(size_t)(rowb + v) * ldc + col] = __float2bfloat16(x);
    }
  }
}

// ---------------- fused gather message GEMM, M-blocked 64x512 ----------------
// m[p,:] = relu( [h[src(p)] | h[dst(p)] | pf[p]] @ Wt_msg^T + b ),  K = 1536.
// Block tile = 64 pairs x 512 cols. Each wave computes 4 m-tiles x 4 n-tiles,
// so every B fragment feeds 4 WMMAs (4x less L2 weight traffic). Gathered A
// rows staged through LDS in 6 K-chunks of 256 (64 x 264 bf16 = 33 KB).
constexpr int AK     = 3 * HID;                 // 1536
constexpr int CH_K   = 256;                     // K-chunk staged in LDS
constexpr int CH_STR = CH_K + 8;                // +16B/row pad -> full bank spread
constexpr int NCHUNK = AK / CH_K;               // 6

__global__ __launch_bounds__(256) void k_msg(
    const __hip_bfloat16* __restrict__ h_bf,
    const __hip_bfloat16* __restrict__ pf_bf,
    const __hip_bfloat16* __restrict__ Wt_msg,
    const float* __restrict__ b_msg,
    __hip_bfloat16* __restrict__ m_bf) {
  __shared__ __hip_bfloat16 Atile[64 * CH_STR]; // 33,792 B

  const int tid  = threadIdx.x;
  const int lane = tid & 31;
  const int wave = tid >> 5;
  const int p0   = blockIdx.x * 64;

  const int kb   = (lane >> 4) * 8;
  const int colb = lane & 15;

  // hoisted B row bases (per n-tile)
  const __hip_bfloat16* Brow[4];
#pragma unroll
  for (int n = 0; n < 4; ++n) {
    const int nt = wave + 8 * n;                // strided n-tile, 0..31
    Brow[n] = Wt_msg + (size_t)(nt * 16 + colb) * AK + kb;
  }

  v8f acc[4][4];
#pragma unroll
  for (int m = 0; m < 4; ++m)
#pragma unroll
    for (int n = 0; n < 4; ++n) acc[m][n] = v8f{};

  for (int ch = 0; ch < NCHUNK; ++ch) {
    const int kbase = ch * CH_K;
    // ---- cooperative gather fill: 2048 x 16B chunks, 8 per thread ----
#pragma unroll
    for (int it = 0; it < 8; ++it) {
      int c   = tid + it * 256;                 // 0..2047
      int row = c >> 5;                         // 32 x 16B chunks per row
      int off = (c & 31) * 8;                   // element offset in chunk, 0..248
      int gk  = kbase + off;                    // element offset in full K row
      // decode pair -> (image, src, dst) for this row
      int p   = p0 + row;
      int img = p / PAIRS_PER;
      int q   = p - img * PAIRS_PER;
      int i   = q / (NPER - 1);
      int r   = q - i * (NPER - 1);
      int j   = r + (r >= i ? 1 : 0);
      const __hip_bfloat16* sp;
      if (gk < HID)           sp = h_bf  + (size_t)(img * NPER + i) * HID + gk;
      else if (gk < 2 * HID)  sp = h_bf  + (size_t)(img * NPER + j) * HID + (gk - HID);
      else                    sp = pf_bf + (size_t)p * HID + (gk - 2 * HID);
      *(v8bf*)&Atile[row * CH_STR + off] = *(const v8bf*)sp;
    }
    __syncthreads();

    // ---- compute on this K-chunk ----
    for (int kk = 0; kk < CH_K; kk += 32) {
      v16bf a[4];
#pragma unroll
      for (int m = 0; m < 4; ++m)
        a[m] = load_frag16(&Atile[(m * 16 + (lane & 15)) * CH_STR + kb] + kk);
#pragma unroll
      for (int n = 0; n < 4; ++n) {
        v16bf b = load_frag16(Brow[n] + kbase + kk);
#pragma unroll
        for (int m = 0; m < 4; ++m)
          acc[m][n] = wmma_bf16(a[m], b, acc[m][n]);
      }
    }
    __syncthreads();
  }

  // ---- epilogue: bias + relu + bf16 store ----
  const int rowoff = 8 * (lane >> 4);
#pragma unroll
  for (int n = 0; n < 4; ++n) {
    const int col = (wave + 8 * n) * 16 + colb;
    const float bv = b_msg[col];
#pragma unroll
    for (int m = 0; m < 4; ++m) {
      const int rowb = p0 + m * 16 + rowoff;
#pragma unroll
      for (int v = 0; v < 8; ++v) {
        float x = fmaxf(acc[m][n][v] + bv, 0.0f);
        m_bf[(size_t)(rowb + v) * HID + col] = __float2bfloat16(x);
      }
    }
  }
}

// ---------------- deterministic segment sum over dst ----------------
__global__ __launch_bounds__(256) void k_segsum(const __hip_bfloat16* __restrict__ m_bf,
                                                float* __restrict__ a_f,
                                                __hip_bfloat16* __restrict__ a_bf) {
  int idx = blockIdx.x * 256 + threadIdx.x;     // node*HID + col
  if (idx >= NNODE * HID) return;
  int node = idx >> 9;
  int col  = idx & (HID - 1);
  int img  = node / NPER;
  int j    = node - img * NPER;
  float s = 0.0f;
#pragma unroll 4
  for (int i = 0; i < NPER; ++i) {
    if (i == j) continue;
    int q = i * (NPER - 1) + (j > i ? j - 1 : j);
    s += __bfloat162float(m_bf[(size_t)(img * PAIRS_PER + q) * HID + col]);
  }
  a_f[idx]  = s;
  a_bf[idx] = __float2bfloat16(s);
}

// ---------------- GRU elementwise ----------------
__global__ __launch_bounds__(256) void k_gru_rh(const float* __restrict__ rpre,
                                                const float* __restrict__ h_f,
                                                __hip_bfloat16* __restrict__ rh_bf,
                                                int n) {
  int i = blockIdx.x * 256 + threadIdx.x;
  if (i >= n) return;
  rh_bf[i] = __float2bfloat16(fast_sigmoid(rpre[i]) * h_f[i]);
}

__global__ __launch_bounds__(256) void k_gru_update(const float* __restrict__ zpre,
                                                    const float* __restrict__ candpre,
                                                    float* __restrict__ h_f,
                                                    __hip_bfloat16* __restrict__ h_bf,
                                                    int n) {
  int i = blockIdx.x * 256 + threadIdx.x;
  if (i >= n) return;
  float z = fast_sigmoid(zpre[i]);
  float c = tanhf(candpre[i]);
  float h = h_f[i];
  h = (1.0f - z) * h + z * c;
  h_f[i]  = h;
  h_bf[i] = __float2bfloat16(h);
}

// ---------------- graph readout (mean over objects) ----------------
__global__ __launch_bounds__(256) void k_readout(const float* __restrict__ objfeat,
                                                 float* __restrict__ gf) {
  int idx = blockIdx.x * 256 + threadIdx.x;     // img*OUT + col
  if (idx >= B_IMG * OUT_DIM) return;
  int img = idx >> 9;
  int col = idx & (OUT_DIM - 1);
  float s = 0.0f;
#pragma unroll 8
  for (int j = 0; j < NPER; ++j)
    s += objfeat[(size_t)(img * NPER + j) * OUT_DIM + col];
  gf[idx] = s * (1.0f / NPER);
}

// ---------------- host side ----------------
static inline size_t al256(size_t x) { return (x + 255) & ~size_t(255); }

extern "C" void kernel_launch(void* const* d_in, const int* in_sizes, int n_in,
                              void* d_out, int out_size, void* d_ws, size_t ws_size,
                              hipStream_t stream) {
  (void)in_sizes; (void)n_in; (void)out_size; (void)ws_size;

  const float* obj_fmaps = (const float*)d_in[1];
  const float* pair_feat = (const float*)d_in[2];
  const float* W_proj = (const float*)d_in[4];
  const float* b_proj = (const float*)d_in[5];
  const float* W_msg  = (const float*)d_in[6];
  const float* b_msg  = (const float*)d_in[7];
  const float* Wz = (const float*)d_in[8];
  const float* Uz = (const float*)d_in[9];
  const float* bz = (const float*)d_in[10];
  const float* Wr = (const float*)d_in[11];
  const float* Ur = (const float*)d_in[12];
  const float* br = (const float*)d_in[13];
  const float* Wh = (const float*)d_in[14];
  const float* Uh = (const float*)d_in[15];
  const float* bh = (const float*)d_in[16];
  const float* W_out = (const float*)d_in[17];
  const float* b_out = (const float*)d_in[18];
  const float* W_cls = (const float*)d_in[19];
  const float* b_cls = (const float*)d_in[20];

  // workspace arena
  char* base = (char*)d_ws;
  size_t off = 0;
  auto alloc = [&](size_t bytes) { char* p = base + off; off = al256(off + bytes); return p; };

  __hip_bfloat16* Wt_proj = (__hip_bfloat16*)alloc((size_t)HID * OBJ_DIM * 2);
  __hip_bfloat16* Wt_msg  = (__hip_bfloat16*)alloc((size_t)HID * 3 * HID * 2);
  __hip_bfloat16* Wt_z    = (__hip_bfloat16*)alloc((size_t)HID * HID * 2);
  __hip_bfloat16* Ut_z    = (__hip_bfloat16*)alloc((size_t)HID * HID * 2);
  __hip_bfloat16* Wt_r    = (__hip_bfloat16*)alloc((size_t)HID * HID * 2);
  __hip_bfloat16* Ut_r    = (__hip_bfloat16*)alloc((size_t)HID * HID * 2);
  __hip_bfloat16* Wt_h    = (__hip_bfloat16*)alloc((size_t)HID * HID * 2);
  __hip_bfloat16* Ut_h    = (__hip_bfloat16*)alloc((size_t)HID * HID * 2);
  __hip_bfloat16* Wt_out  = (__hip_bfloat16*)alloc((size_t)OUT_DIM * HID * 2);
  __hip_bfloat16* Wt_cls  = (__hip_bfloat16*)alloc((size_t)NCLS_PAD * OUT_DIM * 2);
  __hip_bfloat16* fmaps_bf= (__hip_bfloat16*)alloc((size_t)NNODE * OBJ_DIM * 2);
  __hip_bfloat16* pf_bf   = (__hip_bfloat16*)alloc((size_t)NPAIR * HID * 2);
  float*          h_f     = (float*)alloc((size_t)NNODE * HID * 4);
  __hip_bfloat16* h_bf    = (__hip_bfloat16*)alloc((size_t)NNODE * HID * 2);
  __hip_bfloat16* m_bf    = (__hip_bfloat16*)alloc((size_t)NPAIR * HID * 2);
  float*          a_f     = (float*)alloc((size_t)NNODE * HID * 4);
  __hip_bfloat16* a_bf    = (__hip_bfloat16*)alloc((size_t)NNODE * HID * 2);
  float*          zpre    = (float*)alloc((size_t)NNODE * HID * 4);
  float*          rpre    = (float*)alloc((size_t)NNODE * HID * 4);
  float*          candpre = (float*)alloc((size_t)NNODE * HID * 4);
  __hip_bfloat16* rh_bf   = (__hip_bfloat16*)alloc((size_t)NNODE * HID * 2);
  __hip_bfloat16* of_bf   = (__hip_bfloat16*)alloc((size_t)NNODE * OUT_DIM * 2);

  float* out_dists = (float*)d_out;                                  // [1280,151]
  float* out_feat  = out_dists + (size_t)NNODE * NCLS;               // [1280,512]
  float* out_gf    = out_feat + (size_t)NNODE * OUT_DIM;             // [32,512]

  const dim3 blk(256);
  auto gr1 = [](size_t n) { return dim3((unsigned)((n + 255) / 256)); };

  // ---- one-time conversions / weight transposes ----
  k_f32_to_bf16<<<gr1((size_t)NNODE * OBJ_DIM), blk, 0, stream>>>(obj_fmaps, fmaps_bf, NNODE * OBJ_DIM);
  k_f32_to_bf16<<<gr1((size_t)NPAIR * HID), blk, 0, stream>>>(pair_feat, pf_bf, NPAIR * HID);
  k_transpose_bf16<<<gr1((size_t)HID * OBJ_DIM), blk, 0, stream>>>(W_proj, Wt_proj, OBJ_DIM, HID, HID);
  k_transpose_bf16<<<gr1((size_t)HID * 3 * HID), blk, 0, stream>>>(W_msg, Wt_msg, 3 * HID, HID, HID);
  k_transpose_bf16<<<gr1((size_t)HID * HID), blk, 0, stream>>>(Wz, Wt_z, HID, HID, HID);
  k_transpose_bf16<<<gr1((size_t)HID * HID), blk, 0, stream>>>(Uz, Ut_z, HID, HID, HID);
  k_transpose_bf16<<<gr1((size_t)HID * HID), blk, 0, stream>>>(Wr, Wt_r, HID, HID, HID);
  k_transpose_bf16<<<gr1((size_t)HID * HID), blk, 0, stream>>>(Ur, Ut_r, HID, HID, HID);
  k_transpose_bf16<<<gr1((size_t)HID * HID), blk, 0, stream>>>(Wh, Wt_h, HID, HID, HID);
  k_transpose_bf16<<<gr1((size_t)HID * HID), blk, 0, stream>>>(Uh, Ut_h, HID, HID, HID);
  k_transpose_bf16<<<gr1((size_t)OUT_DIM * HID), blk, 0, stream>>>(W_out, Wt_out, HID, OUT_DIM, OUT_DIM);
  k_transpose_bf16<<<gr1((size_t)NCLS_PAD * OUT_DIM), blk, 0, stream>>>(W_cls, Wt_cls, OUT_DIM, NCLS, NCLS_PAD);

  const int NELEM = NNODE * HID;
  const dim3 gElem = gr1((size_t)NELEM);

  // ---- h0 = obj_fmaps @ W_proj + b_proj ----
  k_gemm<4><<<dim3(NNODE / 16), blk, 0, stream>>>(
      fmaps_bf, OBJ_DIM, Wt_proj, OBJ_DIM, OBJ_DIM,
      nullptr, 0, nullptr, 0, 0,
      b_proj, HID, /*act=*/0, h_f, h_bf, HID);

  // ---- T message-passing + GRU steps ----
  for (int t = 0; t < TSTEPS; ++t) {
    k_msg<<<dim3(NPAIR / 64), blk, 0, stream>>>(h_bf, pf_bf, Wt_msg, b_msg, m_bf);
    k_segsum<<<gElem, blk, 0, stream>>>(m_bf, a_f, a_bf);

    k_gemm<4><<<dim3(NNODE / 16), blk, 0, stream>>>(   // z_pre = a@Wz + h@Uz + bz
        a_bf, HID, Wt_z, HID, HID,
        h_bf, HID, Ut_z, HID, HID,
        bz, HID, 0, zpre, nullptr, HID);
    k_gemm<4><<<dim3(NNODE / 16), blk, 0, stream>>>(   // r_pre = a@Wr + h@Ur + br
        a_bf, HID, Wt_r, HID, HID,
        h_bf, HID, Ut_r, HID, HID,
        br, HID, 0, rpre, nullptr, HID);
    k_gru_rh<<<gElem, blk, 0, stream>>>(rpre, h_f, rh_bf, NELEM);
    k_gemm<4><<<dim3(NNODE / 16), blk, 0, stream>>>(   // cand_pre = a@Wh + (r*h)@Uh + bh
        a_bf, HID, Wt_h, HID, HID,
        rh_bf, HID, Ut_h, HID, HID,
        bh, HID, 0, candpre, nullptr, HID);
    k_gru_update<<<gElem, blk, 0, stream>>>(zpre, candpre, h_f, h_bf, NELEM);
  }

  // ---- obj_features = relu(h @ W_out + b_out) ----
  k_gemm<4><<<dim3(NNODE / 16), blk, 0, stream>>>(
      h_bf, HID, Wt_out, HID, HID,
      nullptr, 0, nullptr, 0, 0,
      b_out, OUT_DIM, /*act=*/1, out_feat, of_bf, OUT_DIM);

  // ---- obj_dists = obj_features @ W_cls + b_cls (N=151, B padded to 256) ----
  k_gemm<2><<<dim3(NNODE / 16), blk, 0, stream>>>(
      of_bf, OUT_DIM, Wt_cls, OUT_DIM, OUT_DIM,
      nullptr, 0, nullptr, 0, 0,
      b_cls, NCLS, /*act=*/0, out_dists, nullptr, NCLS);

  // ---- global_features = mean over objects ----
  k_readout<<<gr1((size_t)B_IMG * OUT_DIM), blk, 0, stream>>>(out_feat, out_gf);
}
